// TemporalAttention_39333310497276
// MI455X (gfx1250) — matrix operand order repaired and checked by reading, not compiled, not run on hardware
//
#include <hip/hip_runtime.h>
#include <stdint.h>

// Problem sizes (match reference)
#define Hd 1024
#define Sd 4096
#define Bd 4
#define NROW (Bd*Sd)      // 16384 total rows (b*S+s)
#define QT  (NROW/16)     // 1024 16-row query tiles
#define DT  (Hd/16)       // 64 feature tiles
#define KT  (Sd/16)       // 256 key tiles per batch
#define PPART 8           // softmax-stat key partitions
#define QSTR 1032         // padded LDS row stride (bf16 elems) for staged Q tiles

typedef __attribute__((ext_vector_type(16))) __bf16     v16bf;
typedef __attribute__((ext_vector_type(8)))  float      v8f;
typedef __attribute__((ext_vector_type(4)))  unsigned   u32x4;
typedef __attribute__((ext_vector_type(4)))  float      f32x4;

union FragB { u32x4 q[2]; unsigned u[8]; v16bf v; };   // 32 B: one 16x32 A or 32x16 B bf16 fragment per lane

static __device__ __forceinline__ unsigned short bfbits(float f) {
  union { float f; unsigned u; } c; c.f = f;
  return (unsigned short)((c.u + 0x8000u) >> 16);      // round-half-up f32 -> bf16
}
// pack two f32 -> packed bf16 pair: 2x v_add + 1x v_perm_b32
static __device__ __forceinline__ unsigned pk2(float lo, float hi) {
  union { float f; unsigned u; } a, b; a.f = lo; b.f = hi;
  return __builtin_amdgcn_perm(b.u + 0x8000u, a.u + 0x8000u, 0x07060302);
}
static __device__ __forceinline__ v8f wmma_bf16(const FragB& a, const FragB& b, v8f c) {
  return __builtin_amdgcn_wmma_f32_16x16x32_bf16(false, a.v, false, b.v, (short)0, c, false, false);
}

// ---------------------------------------------------------------------------
// K0: transpose-convert Wq/Wk/Wv -> Wt[m][d][h] = W_m[h][d] in bf16
// ---------------------------------------------------------------------------
__global__ void k_prep_w(const float* __restrict__ Wq, const float* __restrict__ Wk,
                         const float* __restrict__ Wv, unsigned short* __restrict__ Wt) {
  size_t t = (size_t)blockIdx.x * blockDim.x + threadIdx.x;   // 3*H*H threads
  int m = (int)(t / ((size_t)Hd * Hd));
  size_t r = t % ((size_t)Hd * Hd);
  int d = (int)(r / Hd), h = (int)(r % Hd);
  const float* W = (m == 0) ? Wq : (m == 1) ? Wk : Wv;
  Wt[t] = bfbits(W[(size_t)h * Hd + d]);
}

// ---------------------------------------------------------------------------
// K1: QKV projection. One wave per 16x64 output strip (4 independent WMMA
// accumulator chains, A fragment built once per k-step).
// ---------------------------------------------------------------------------
__global__ void k_qkv(const float* __restrict__ x, const unsigned short* __restrict__ Wt,
                      const float* __restrict__ bq, const float* __restrict__ bk,
                      const float* __restrict__ bv,
                      unsigned short* __restrict__ Qb, unsigned short* __restrict__ Kb,
                      unsigned short* __restrict__ Vt) {
  int wave = blockIdx.x * (blockDim.x >> 5) + (threadIdx.x >> 5);
  int lane = threadIdx.x & 31, hid = lane >> 4, l16 = lane & 15;
  int m   = wave / (QT * (DT / 4));
  int r2  = wave % (QT * (DT / 4));
  int st  = r2 / (DT / 4);
  int dt0 = (r2 % (DT / 4)) * 4;
  int arow = st * 16 + l16;
  const float* xr = x + (size_t)arow * Hd;

  int dcol[4];
  const unsigned short* Wrow[4];
#pragma unroll
  for (int j = 0; j < 4; j++) {
    dcol[j] = (dt0 + j) * 16 + l16;
    Wrow[j] = Wt + ((size_t)m * Hd + dcol[j]) * (size_t)Hd;
  }

  v8f zero = {};
  v8f acc[4] = { zero, zero, zero, zero };
  for (int k0 = 0; k0 < Hd; k0 += 32) {
    __builtin_prefetch(xr + k0 + 128, 0, 0);          // stream x ahead
    FragB a;
    const f32x4* xa  = (const f32x4*)(xr + k0 + 8 * hid);
    const f32x4* xa2 = (const f32x4*)(xr + k0 + 16 + 8 * hid);
    f32x4 f0 = xa[0],  f1 = xa[1];
    f32x4 f2 = xa2[0], f3 = xa2[1];
    a.u[0] = pk2(f0.x, f0.y); a.u[1] = pk2(f0.z, f0.w);
    a.u[2] = pk2(f1.x, f1.y); a.u[3] = pk2(f1.z, f1.w);
    a.u[4] = pk2(f2.x, f2.y); a.u[5] = pk2(f2.z, f2.w);
    a.u[6] = pk2(f3.x, f3.y); a.u[7] = pk2(f3.z, f3.w);
#pragma unroll
    for (int j = 0; j < 4; j++) {
      FragB bfr;
      const u32x4* bp = (const u32x4*)(Wrow[j] + k0 + 8 * hid);
      bfr.q[0] = bp[0]; bfr.q[1] = bp[2];
      acc[j] = wmma_bf16(a, bfr, acc[j]);
    }
  }
  const float* bias = (m == 0) ? bq : (m == 1) ? bk : bv;
#pragma unroll
  for (int j = 0; j < 4; j++) {
    float bl = bias[dcol[j]];
#pragma unroll
    for (int r = 0; r < 8; r++) acc[j][r] += bl;
  }

  if (m == 2) {                           // V -> transposed Vt[b][d][s]
    int bb = (st * 16) >> 12;
#pragma unroll
    for (int j = 0; j < 4; j++)
#pragma unroll
      for (int r = 0; r < 8; r++) {
        int row = st * 16 + r + 8 * hid;
        int s   = row & (Sd - 1);
        Vt[((size_t)bb * Hd + dcol[j]) * (size_t)Sd + s] = bfbits(acc[j][r]);
      }
  } else {
    unsigned short* dst = (m == 0) ? Qb : Kb;
#pragma unroll
    for (int j = 0; j < 4; j++)
#pragma unroll
      for (int r = 0; r < 8; r++) {
        int row = st * 16 + r + 8 * hid;
        dst[(size_t)row * Hd + dcol[j]] = bfbits(acc[j][r]);
      }
  }
}

// Stage one 16x1024 bf16 Q tile (32 KB) into LDS with async global->LDS copies.
static __device__ __forceinline__ void stage_qtile(const unsigned short* Qb, int qrow0,
                                                   unsigned ldsbase, int tid) {
#pragma unroll
  for (int s = 0; s < 8; s++) {
    int i = s * 256 + tid;
    int row = i >> 7, seg = i & 127;                   // 128 x 16 B segments per row
    unsigned ldsoff = ldsbase + row * (QSTR * 2) + seg * 16;
    unsigned long long ga =
        (unsigned long long)(size_t)(Qb + (size_t)(qrow0 + row) * Hd + seg * 8);
    asm volatile("global_load_async_to_lds_b128 %0, %1, off"
                 :: "v"(ldsoff), "v"(ga) : "memory");
  }
  asm volatile("s_wait_asynccnt 0x0" ::: "memory");
  __syncthreads();
}

// ---------------------------------------------------------------------------
// K2a: softmax statistics. Block = 8 waves sharing one async-staged Q tile
// in LDS; wave w owns key partition w. 4 key tiles per pass (4 WMMA chains).
// ---------------------------------------------------------------------------
__global__ void __launch_bounds__(256)
k_stats(const unsigned short* __restrict__ Qb,
        const unsigned short* __restrict__ Kb,
        const float* __restrict__ scale_p,
        float* __restrict__ partM, float* __restrict__ partL) {
  __shared__ __attribute__((aligned(16))) unsigned short Qlds[16 * QSTR];
  int tid = threadIdx.x;
  int wv = tid >> 5, lane = tid & 31, hid = lane >> 4, l16 = lane & 15;
  int qt = blockIdx.x, p = wv;
  float inv_scale = 1.0f / scale_p[0];
  int bb = (qt * 16) >> 12;
  unsigned ldsbase = (unsigned)(size_t)(&Qlds[0]);     // LDS byte offset = low 32 bits of flat addr
  stage_qtile(Qb, qt * 16, ldsbase, tid);

  const unsigned short* Ql = &Qlds[l16 * QSTR];
  float run_m[8], run_l[8];
#pragma unroll
  for (int r = 0; r < 8; r++) { run_m[r] = -3.0e38f; run_l[r] = 0.f; }

  v8f zero = {};
  for (int kt = p * (KT / PPART); kt < (p + 1) * (KT / PPART); kt += 4) {
    const unsigned short* Krow[4];
#pragma unroll
    for (int j = 0; j < 4; j++)
      Krow[j] = Kb + (size_t)(bb * Sd + (kt + j) * 16 + l16) * Hd;
    v8f acc[4] = { zero, zero, zero, zero };
    for (int h0 = 0; h0 < Hd; h0 += 32) {
      FragB a;
      a.q[0] = *(const u32x4*)(Ql + h0 + 8 * hid);
      a.q[1] = *(const u32x4*)(Ql + h0 + 16 + 8 * hid);
#pragma unroll
      for (int j = 0; j < 4; j++) {
        FragB bfr;
        const u32x4* bp = (const u32x4*)(Krow[j] + h0 + 8 * hid);
        bfr.q[0] = bp[0]; bfr.q[1] = bp[2];
        acc[j] = wmma_bf16(a, bfr, acc[j]);
      }
    }
#pragma unroll
    for (int j = 0; j < 4; j++) {
#pragma unroll
      for (int r = 0; r < 8; r++) {
        float s = acc[j][r] * inv_scale;
        float mx = s;                                 // row max over 16-lane half-group
        mx = fmaxf(mx, __shfl_xor(mx, 1, 32));
        mx = fmaxf(mx, __shfl_xor(mx, 2, 32));
        mx = fmaxf(mx, __shfl_xor(mx, 4, 32));
        mx = fmaxf(mx, __shfl_xor(mx, 8, 32));
        float nm = fmaxf(run_m[r], mx);
        float e = __expf(s - nm);
        e += __shfl_xor(e, 1, 32);
        e += __shfl_xor(e, 2, 32);
        e += __shfl_xor(e, 4, 32);
        e += __shfl_xor(e, 8, 32);
        run_l[r] = run_l[r] * __expf(run_m[r] - nm) + e;
        run_m[r] = nm;
      }
    }
  }
  if (l16 == 0) {
#pragma unroll
    for (int r = 0; r < 8; r++) {
      int q = qt * 16 + r + 8 * hid;
      partM[(size_t)q * PPART + p] = run_m[r];
      partL[(size_t)q * PPART + p] = run_l[r];
    }
  }
}

// K2r: combine partial stats -> final max M and reciprocal normalizer 1/L
__global__ void k_reduce(const float* __restrict__ partM, const float* __restrict__ partL,
                         float* __restrict__ Mfin, float* __restrict__ Linv) {
  int q = blockIdx.x * blockDim.x + threadIdx.x;
  float m = -3.0e38f;
#pragma unroll
  for (int p = 0; p < PPART; p++) m = fmaxf(m, partM[(size_t)q * PPART + p]);
  float l = 0.f;
#pragma unroll
  for (int p = 0; p < PPART; p++) l += partL[(size_t)q * PPART + p] * __expf(partM[(size_t)q * PPART + p] - m);
  Mfin[q] = m; Linv[q] = 1.0f / l;
}

// ---------------------------------------------------------------------------
// K2b: recompute scores, softmax with known (M, 1/L), write P bf16.
// Block = 8 waves sharing the async-staged Q tile; wave owns 16 key tiles, 4 per pass.
// ---------------------------------------------------------------------------
__global__ void __launch_bounds__(256)
k_probs(const unsigned short* __restrict__ Qb,
        const unsigned short* __restrict__ Kb,
        const float* __restrict__ scale_p,
        const float* __restrict__ Mfin, const float* __restrict__ Linv,
        unsigned short* __restrict__ P) {
  __shared__ __attribute__((aligned(16))) unsigned short Qlds[16 * QSTR];
  int tid = threadIdx.x;
  int wv = tid >> 5, lane = tid & 31, hid = lane >> 4, l16 = lane & 15;
  const int KC = 16;                                  // key tiles per wave
  int qt = blockIdx.x >> 1;
  int kc = (blockIdx.x & 1) * 8 + wv;                 // 16 chunks per query tile
  float inv_scale = 1.0f / scale_p[0];
  int bb = (qt * 16) >> 12;
  unsigned ldsbase = (unsigned)(size_t)(&Qlds[0]);
  stage_qtile(Qb, qt * 16, ldsbase, tid);

  const unsigned short* Ql = &Qlds[l16 * QSTR];
  float m8[8], li8[8];
#pragma unroll
  for (int r = 0; r < 8; r++) {
    int q = qt * 16 + r + 8 * hid;
    m8[r] = Mfin[q]; li8[r] = Linv[q];
  }

  v8f zero = {};
  for (int kt = kc * KC; kt < kc * KC + KC; kt += 4) {
    const unsigned short* Krow[4];
#pragma unroll
    for (int j = 0; j < 4; j++)
      Krow[j] = Kb + (size_t)(bb * Sd + (kt + j) * 16 + l16) * Hd;
    v8f acc[4] = { zero, zero, zero, zero };
    for (int h0 = 0; h0 < Hd; h0 += 32) {
      FragB a;
      a.q[0] = *(const u32x4*)(Ql + h0 + 8 * hid);
      a.q[1] = *(const u32x4*)(Ql + h0 + 16 + 8 * hid);
#pragma unroll
      for (int j = 0; j < 4; j++) {
        FragB bfr;
        const u32x4* bp = (const u32x4*)(Krow[j] + h0 + 8 * hid);
        bfr.q[0] = bp[0]; bfr.q[1] = bp[2];
        acc[j] = wmma_bf16(a, bfr, acc[j]);
      }
    }
#pragma unroll
    for (int j = 0; j < 4; j++)
#pragma unroll
      for (int r = 0; r < 8; r++) {
        float pv = __expf(acc[j][r] * inv_scale - m8[r]) * li8[r];
        int qg  = qt * 16 + r + 8 * hid;
        int key = (kt + j) * 16 + l16;
        P[(size_t)qg * Sd + key] = bfbits(pv);
      }
  }
}

// ---------------------------------------------------------------------------
// K3: O = P @ V. Block = 8 waves sharing one 16-query row-block of P.
// Double-buffered async global->LDS staging (s_wait_asynccnt 2) overlaps the
// next 8 KB P chunk with WMMA on the current one; each wave owns 4 d-tiles.
// ---------------------------------------------------------------------------
__global__ void __launch_bounds__(256)
k_out(const unsigned short* __restrict__ P,
      const unsigned short* __restrict__ Vt,
      float* __restrict__ out) {
  __shared__ __attribute__((aligned(16))) unsigned short Plds[2][16 * 264]; // 2 x (16 rows x (256+8))
  int tid = threadIdx.x;
  int wv = tid >> 5, lane = tid & 31, hid = lane >> 4, l16 = lane & 15;
  int qt = blockIdx.x >> 1;                  // 1024 query tiles
  int dg = blockIdx.x & 1;                   // which half of the 64 feature tiles
  int bb = (qt * 16) >> 12;
  int qrow0 = qt * 16;

  int dt0 = dg * 32 + wv * 4;                // 4 d-tiles per wave
  int dcol[4];
  const unsigned short* Vrow[4];
#pragma unroll
  for (int j = 0; j < 4; j++) {
    dcol[j] = (dt0 + j) * 16 + l16;
    Vrow[j] = Vt + ((size_t)bb * Hd + dcol[j]) * (size_t)Sd;
  }
  unsigned ldsbase = (unsigned)(size_t)(&Plds[0][0]);

  auto stage = [&](int k0, int bufi) {
#pragma unroll
    for (int s = 0; s < 2; s++) {
      int i = s * 256 + tid;
      int row = i >> 5, seg = i & 31;
      unsigned ldsoff = ldsbase + bufi * (16 * 264 * 2) + row * 528 + seg * 16;
      unsigned long long ga =
          (unsigned long long)(size_t)(P + (size_t)(qrow0 + row) * Sd + k0 + seg * 8);
      asm volatile("global_load_async_to_lds_b128 %0, %1, off"
                   :: "v"(ldsoff), "v"(ga) : "memory");
    }
  };

  v8f zero = {};
  v8f acc[4] = { zero, zero, zero, zero };
  const int NC = Sd / 256;                   // 16 chunks
  stage(0, 0);
  for (int c = 0; c < NC; c++) {
    if (c + 1 < NC) {
      stage((c + 1) * 256, (c + 1) & 1);
      asm volatile("s_wait_asynccnt 0x2" ::: "memory");  // older chunk complete (in-order)
    } else {
      asm volatile("s_wait_asynccnt 0x0" ::: "memory");
    }
    __syncthreads();
    const unsigned short* Pl = &Plds[c & 1][0];
    int k0 = c * 256;
    for (int kl = 0; kl < 256; kl += 32) {
      FragB a;
      a.q[0] = *(const u32x4*)(Pl + l16 * 264 + kl + 8 * hid);
      a.q[1] = *(const u32x4*)(Pl + l16 * 264 + kl + 16 + 8 * hid);
      int kg = k0 + kl + 8 * hid;
#pragma unroll
      for (int j = 0; j < 4; j++) {
        FragB bfr;
        const u32x4* bp = (const u32x4*)(Vrow[j] + kg);
        bfr.q[0] = bp[0]; bfr.q[1] = bp[2];
        acc[j] = wmma_bf16(a, bfr, acc[j]);
      }
    }
    __syncthreads();                         // readers done before buffer is re-staged
  }
#pragma unroll
  for (int j = 0; j < 4; j++)
#pragma unroll
    for (int r = 0; r < 8; r++) {
      int row = qrow0 + r + 8 * hid;
      out[(size_t)row * Hd + dcol[j]] = acc[j][r];
    }
}

// ---------------------------------------------------------------------------
extern "C" void kernel_launch(void* const* d_in, const int* in_sizes, int n_in,
                              void* d_out, int out_size, void* d_ws, size_t ws_size,
                              hipStream_t stream) {
  const float* x  = (const float*)d_in[0];
  const float* Wq = (const float*)d_in[1];
  const float* bq = (const float*)d_in[2];
  const float* Wk = (const float*)d_in[3];
  const float* bk = (const float*)d_in[4];
  const float* Wv = (const float*)d_in[5];
  const float* bv = (const float*)d_in[6];
  const float* sc = (const float*)d_in[7];
  float* out = (float*)d_out;

  char* ws = (char*)d_ws;
  size_t off = 0;
  unsigned short* Wt = (unsigned short*)(ws + off); off += (size_t)3 * Hd * Hd * 2;   //   6 MB
  unsigned short* Qb = (unsigned short*)(ws + off); off += (size_t)NROW * Hd * 2;     //  32 MB
  unsigned short* Kb = (unsigned short*)(ws + off); off += (size_t)NROW * Hd * 2;     //  32 MB
  unsigned short* Vt = (unsigned short*)(ws + off); off += (size_t)Bd * Hd * Sd * 2;  //  32 MB
  float* partM = (float*)(ws + off); off += (size_t)NROW * PPART * 4;
  float* partL = (float*)(ws + off); off += (size_t)NROW * PPART * 4;
  float* Mfin  = (float*)(ws + off); off += (size_t)NROW * 4;
  float* Linv  = (float*)(ws + off); off += (size_t)NROW * 4;
  unsigned short* P = (unsigned short*)(ws + off); off += (size_t)NROW * Sd * 2;      // 128 MB

  k_prep_w<<<3 * Hd * Hd / 256, 256, 0, stream>>>(Wq, Wk, Wv, Wt);
  k_qkv   <<<3 * QT * (DT/4) / 8, 256, 0, stream>>>(x, Wt, bq, bk, bv, Qb, Kb, Vt);
  k_stats <<<QT,                  256, 0, stream>>>(Qb, Kb, sc, partM, partL);
  k_reduce<<<NROW / 256,          256, 0, stream>>>(partM, partL, Mfin, Linv);
  k_probs <<<QT * 2,              256, 0, stream>>>(Qb, Kb, sc, Mfin, Linv, P);
  k_out   <<<QT * 2,              256, 0, stream>>>(P, Vt, out);
}
